// CONV_tiny_add_partial_60138132079272
// MI455X (gfx1250) — matrix operand order repaired and checked by reading, not compiled, not run on hardware
//
#include <hip/hip_runtime.h>

// ---------------------------------------------------------------------------
// CDNA5 (gfx1250) fused CNN inference:
//   conv5x5+BN+ReLU+avgpool2+clip  x3   (implicit GEMM on v_wmma_f32_16x16x32_f16)
//   + final 4x4-conv FC + BN1d affine   (WMMA GEMM)
// One wave owns one 4x4 spatial patch of one image and ALL cout tiles (MT),
// so the staged LDS patch + B-gather feed MT WMMAs per K-chunk.
// ---------------------------------------------------------------------------

typedef _Float16 f16;
typedef __attribute__((ext_vector_type(16))) _Float16 v16h;
typedef __attribute__((ext_vector_type(8)))  float    v8f;

#define EPSBN 1e-5f

// -------------------------------------------------------------------- prep --
__global__ void k_cast_f32_to_f16(const float* __restrict__ in,
                                  f16* __restrict__ out, int n) {
  for (int i = blockIdx.x * blockDim.x + threadIdx.x; i < n;
       i += gridDim.x * blockDim.x)
    out[i] = (f16)in[i];
}

__global__ void k_bn_fold(const float* __restrict__ g, const float* __restrict__ b,
                          const float* __restrict__ m, const float* __restrict__ v,
                          float* __restrict__ scale, float* __restrict__ bias,
                          int C, int Cpad) {
  int i = blockIdx.x * blockDim.x + threadIdx.x;
  if (i >= Cpad) return;
  if (i < C) {
    float s  = g[i] * rsqrtf(v[i] + EPSBN);
    scale[i] = s;
    bias[i]  = b[i] - m[i] * s;
  } else {
    scale[i] = 0.f;
    bias[i]  = 0.f;
  }
}

// Pack weights into the exact per-lane WMMA A-operand order.
// Layout: out[((mt*KC + kc)*32 + lane)*16 + j]
// A 16x32 f16 layout (contiguous, matches B): lane<16 -> K in {0..15},
//                                  lane>=16 -> K in {16..31}
// conv5x5 mode: K = ci*32 + t, taps t<25 valid (25 taps padded to 32 per Cin).
// direct mode (FC): K indexes the flat [Cin*4*4] patch directly.
__global__ void k_pack_w(const float* __restrict__ w, f16* __restrict__ out,
                         int Cout, int Cin, int KC, int mtiles, int conv5x5) {
  int total = mtiles * KC * 32 * 16;
  for (int i = blockIdx.x * blockDim.x + threadIdx.x; i < total;
       i += gridDim.x * blockDim.x) {
    int j    = i & 15;
    int lane = (i >> 4) & 31;
    int rest = i >> 9;
    int kc   = rest % KC;
    int mt   = rest / KC;
    int m    = mt * 16 + (lane & 15);
    int hi   = lane >> 4;
    int kk   = hi * 16 + j;
    int K    = kc * 32 + kk;
    float val = 0.f;
    if (m < Cout) {
      if (conv5x5) {
        int t = K & 31, ci = K >> 5;
        if (t < 25) val = w[((size_t)m * Cin + ci) * 25 + t];
      } else {
        val = w[(size_t)m * (KC * 32) + K];
      }
    }
    out[i] = (f16)val;
  }
}

// ------------------------------------------------------- fused conv block --
// MT = number of 16-cout tiles handled by one wave (Cout = MT*16).
// sxs/sys = log2 of spatial tile grid (all power-of-two here).
template <int MT>
__global__ __launch_bounds__(256) void k_conv_wmma(
    const f16* __restrict__ actin, const f16* __restrict__ wpack,
    const float* __restrict__ scale, const float* __restrict__ bias,
    f16* __restrict__ actout, int Cin, int H, int W,
    int sxs, int sys, int total_waves) {
  extern __shared__ char smem[];
  const int lane = threadIdx.x & 31;
  const int wid  = threadIdx.x >> 5;
  const int gw   = blockIdx.x * 8 + wid;
  if (gw >= total_waves) return;

  const int patch_halfs = Cin * 72;            // per-ci: 8x8 patch + 8 zero slots
  f16*   patch = (f16*)smem + (size_t)wid * (patch_halfs + 512);
  float* ctile = (float*)(patch + patch_halfs);    // 16x16 f32 epilogue tile

  // decode wave -> (image, spatial tile) with shifts/masks only
  const int sx = gw & ((1 << sxs) - 1);
  const int sy = (gw >> sxs) & ((1 << sys) - 1);
  const int n  = gw >> (sxs + sys);
  const int h0 = sy * 4, w0 = sx * 4;

  // ---- stage 8x8 (+zero pad) input patch per channel into LDS -------------
  const f16* img = actin + (size_t)n * Cin * H * W;
  for (int ci = 0; ci < Cin; ++ci) {
    const f16* ip = img + (size_t)ci * H * W;
    f16* pp = patch + ci * 72;
    for (int t = lane; t < 72; t += 32) {
      f16 v = (f16)0;
      if (t < 64) {
        int hh = h0 - 2 + (t >> 3);
        int ww = w0 - 2 + (t & 7);
        if (hh >= 0 && hh < H && ww >= 0 && ww < W) v = ip[hh * W + ww];
      }
      pp[t] = v;
    }
  }
  asm volatile("s_wait_dscnt 0" ::: "memory");

  // ---- fixed per-lane B-gather offsets (B layout: lane<16 K=j, lane>=16 K=16+j)
  const int  p  = lane & 15;       // pixel column of B / C
  const int  r  = p >> 2, c = p & 3;
  const bool hi = lane >= 16;
  int boff[16];
#pragma unroll
  for (int j = 0; j < 16; ++j) {
    int t = (hi ? 16 : 0) + j;                       // tap index within chunk
    boff[j] = (t < 25) ? ((r + t / 5) * 8 + (c + t % 5))   // 5x5 tap
                       : (64 + (t - 25));                  // zeroed pad slot
  }

  // ---- K loop: one WMMA chunk per input channel, MT cout-tiles per chunk --
  v8f acc[MT];
#pragma unroll
  for (int mt = 0; mt < MT; ++mt) acc[mt] = (v8f){};

  const v16h* wp = (const v16h*)wpack + lane;
  for (int kc = 0; kc < Cin; ++kc) {
    const f16* pb = patch + kc * 72;
    v16h b;
#pragma unroll
    for (int j = 0; j < 16; ++j) b[j] = pb[boff[j]];
#pragma unroll
    for (int mt = 0; mt < MT; ++mt) {
      v16h a = wp[((size_t)mt * Cin + kc) * 32];     // packed A strip, 32B load
      acc[mt] = __builtin_amdgcn_wmma_f32_16x16x32_f16(
          false, a, false, b, (short)0, acc[mt], false, false);
    }
  }

  // ---- epilogue per cout tile: BN+ReLU -> LDS, 2x2 avgpool + clip -> f16 --
  const int Hp = H >> 1, Wp = W >> 1;
  const int Cout = MT * 16;
  const int mrow = (lane >> 4) * 8;   // C layout: vgpr r -> M=r (lo) / 8+r (hi)
  const int m    = lane & 15;
  const int grp  = lane >> 4;
#pragma unroll
  for (int mt = 0; mt < MT; ++mt) {
#pragma unroll
    for (int rr = 0; rr < 8; ++rr) {
      int gm  = mt * 16 + mrow + rr;
      float v = acc[mt][rr] * scale[gm] + bias[gm];
      ctile[(mrow + rr) * 16 + p] = fmaxf(v, 0.0f);
    }
    asm volatile("s_wait_dscnt 0" ::: "memory");
#pragma unroll
    for (int q = 0; q < 2; ++q) {
      int pp_ = grp * 2 + q;
      int pr = pp_ >> 1, pc = pp_ & 1;
      const float* row = ctile + m * 16;
      float s = row[(2 * pr) * 4 + 2 * pc]     + row[(2 * pr) * 4 + 2 * pc + 1] +
                row[(2 * pr + 1) * 4 + 2 * pc] + row[(2 * pr + 1) * 4 + 2 * pc + 1];
      s = fminf(fmaxf(s * 0.25f, 0.0f), 1.0f);
      int gm = mt * 16 + m;
      size_t oidx =
          (((size_t)n * Cout + gm) * Hp + (sy * 2 + pr)) * Wp + (sx * 2 + pc);
      actout[oidx] = (f16)s;
    }
    // LDS ops are in-order within a wave: next tile's ctile stores cannot
    // pass this tile's pool loads, so no extra wait is needed here.
  }
}

// ------------------------------------------------------------ FC as GEMM --
// M=16 (10 padded), N=B images, K=1024 = flat [64*4*4] activation.
__global__ __launch_bounds__(256) void k_fc_wmma(
    const f16* __restrict__ act3, const f16* __restrict__ wpack,
    const float* __restrict__ scale, const float* __restrict__ bias,
    float* __restrict__ out, int ntiles) {
  const int lane = threadIdx.x & 31;
  const int gw   = blockIdx.x * 8 + (threadIdx.x >> 5);
  if (gw >= ntiles) return;

  const int  p  = lane & 15;
  const bool hi = lane >= 16;
  const int  n  = gw * 16 + p;
  const f16* bbase = act3 + (size_t)n * 1024 + (hi ? 16 : 0);
  const v16h* wp   = (const v16h*)wpack + lane;

  v8f acc = {};
  for (int kc = 0; kc < 32; ++kc) {
    v16h a = wp[(size_t)kc * 32];
    v16h b = *(const v16h*)(bbase + kc * 32);   // contiguous 32B -> b128 x2
    acc = __builtin_amdgcn_wmma_f32_16x16x32_f16(
        false, a, false, b, (short)0, acc, false, false);
  }
  const int mrow = (lane >> 4) * 8;
#pragma unroll
  for (int rr = 0; rr < 8; ++rr) {
    int m = mrow + rr;
    if (m < 10) out[(size_t)n * 10 + m] = acc[rr] * scale[m] + bias[m];
  }
}

// ------------------------------------------------------------------ launch --
extern "C" void kernel_launch(void* const* d_in, const int* in_sizes, int n_in,
                              void* d_out, int out_size, void* d_ws, size_t ws_size,
                              hipStream_t stream) {
  (void)n_in; (void)out_size; (void)ws_size;
  const float* x  = (const float*)d_in[0];
  const float* w1 = (const float*)d_in[1];
  const float* g1 = (const float*)d_in[2],  *b1 = (const float*)d_in[3];
  const float* m1 = (const float*)d_in[4],  *v1 = (const float*)d_in[5];
  const float* w2 = (const float*)d_in[6];
  const float* g2 = (const float*)d_in[7],  *b2 = (const float*)d_in[8];
  const float* m2 = (const float*)d_in[9],  *v2 = (const float*)d_in[10];
  const float* w3 = (const float*)d_in[11];
  const float* g3 = (const float*)d_in[12], *b3 = (const float*)d_in[13];
  const float* m3 = (const float*)d_in[14], *v3 = (const float*)d_in[15];
  const float* w4 = (const float*)d_in[16];
  const float* g4 = (const float*)d_in[17], *b4 = (const float*)d_in[18];
  const float* m4 = (const float*)d_in[19], *v4 = (const float*)d_in[20];
  float* out = (float*)d_out;

  const int B = in_sizes[0] / (3 * 32 * 32);

  // workspace carve-out (256B aligned)
  char* ws = (char*)d_ws;
  size_t off = 0;
  auto take = [&](size_t bytes) {
    void* pp = ws + off;
    off = (off + bytes + 255) & ~(size_t)255;
    return pp;
  };
  f16* act0 = (f16*)take((size_t)B * 3  * 1024 * 2);
  f16* act1 = (f16*)take((size_t)B * 32 * 256  * 2);
  f16* act2 = (f16*)take((size_t)B * 32 * 64   * 2);
  f16* act3 = (f16*)take((size_t)B * 64 * 16   * 2);
  f16* wp1  = (f16*)take((size_t)2 * 3  * 512 * 2);
  f16* wp2  = (f16*)take((size_t)2 * 32 * 512 * 2);
  f16* wp3  = (f16*)take((size_t)4 * 32 * 512 * 2);
  f16* wp4  = (f16*)take((size_t)1 * 32 * 512 * 2);
  float* s1 = (float*)take(32 * 4); float* bb1 = (float*)take(32 * 4);
  float* s2 = (float*)take(32 * 4); float* bb2 = (float*)take(32 * 4);
  float* s3 = (float*)take(64 * 4); float* bb3 = (float*)take(64 * 4);
  float* s4 = (float*)take(16 * 4); float* bb4 = (float*)take(16 * 4);

  // ---- prep ----------------------------------------------------------------
  int ncast = B * 3 * 1024;
  int cblk  = (ncast + 255) / 256;
  k_cast_f32_to_f16<<<cblk < 2048 ? cblk : 2048, 256, 0, stream>>>(x, act0, ncast);
  k_bn_fold<<<1, 64, 0, stream>>>(g1, b1, m1, v1, s1, bb1, 32, 32);
  k_bn_fold<<<1, 64, 0, stream>>>(g2, b2, m2, v2, s2, bb2, 32, 32);
  k_bn_fold<<<1, 64, 0, stream>>>(g3, b3, m3, v3, s3, bb3, 64, 64);
  k_bn_fold<<<1, 64, 0, stream>>>(g4, b4, m4, v4, s4, bb4, 10, 16);
  k_pack_w<<<64,  256, 0, stream>>>(w1, wp1, 32, 3,  3,  2, 1);
  k_pack_w<<<128, 256, 0, stream>>>(w2, wp2, 32, 32, 32, 2, 1);
  k_pack_w<<<256, 256, 0, stream>>>(w3, wp3, 64, 32, 32, 4, 1);
  k_pack_w<<<64,  256, 0, stream>>>(w4, wp4, 10, 64, 32, 1, 0);

  // ---- fused conv blocks (one wave: 4x4 patch x all cout tiles) -----------
  // layer 1: Cin=3, 32x32 -> pooled 16x16, Cout=32 (MT=2), 8x8 spatial tiles
  {
    int waves = B * 8 * 8;
    int smem  = 8 * ((3 * 72 + 512) * 2);
    k_conv_wmma<2><<<(waves + 7) / 8, 256, smem, stream>>>(
        act0, wp1, s1, bb1, act1, 3, 32, 32, 3, 3, waves);
  }
  // layer 2: Cin=32, 16x16 -> pooled 8x8, Cout=32 (MT=2), 4x4 spatial tiles
  {
    int waves = B * 4 * 4;
    int smem  = 8 * ((32 * 72 + 512) * 2);
    k_conv_wmma<2><<<(waves + 7) / 8, 256, smem, stream>>>(
        act1, wp2, s2, bb2, act2, 32, 16, 16, 2, 2, waves);
  }
  // layer 3: Cin=32, 8x8 -> pooled 4x4, Cout=64 (MT=4), 2x2 spatial tiles
  {
    int waves = B * 2 * 2;
    int smem  = 8 * ((32 * 72 + 512) * 2);
    k_conv_wmma<4><<<(waves + 7) / 8, 256, smem, stream>>>(
        act2, wp3, s3, bb3, act3, 32, 8, 8, 1, 1, waves);
  }
  // ---- FC + BN1d -----------------------------------------------------------
  {
    int ntiles = B / 16;
    k_fc_wmma<<<(ntiles + 7) / 8, 256, 0, stream>>>(act3, wp4, s4, bb4, out,
                                                    ntiles);
  }
}
